// RetouchGenerator_81209241633611
// MI455X (gfx1250) — matrix operand order repaired and checked by reading, not compiled, not run on hardware
//
#include <hip/hip_runtime.h>

typedef __attribute__((ext_vector_type(16))) _Float16 v16h;
typedef __attribute__((ext_vector_type(8)))  float    v8f;
typedef __attribute__((ext_vector_type(4)))  unsigned v4u;
typedef __attribute__((ext_vector_type(8)))  int      v8i;
typedef __attribute__((ext_vector_type(4)))  int      v4i;

#define HRDIM 1024
#define NPIX  (HRDIM * HRDIM)
#define GRIDN 24576              // 12*16*16*8 floats = 96 KB bilateral grid
#define NPAR  109                // 9 pw_mat + 3 pw_bias_tag + 1 pw_bias + 48 rho_a + 48 rho_t

// ---------------------------------------------------------------------------
// Direct 3x3 conv + ReLU for the tiny-channel layers (ll1..ll3).
// ---------------------------------------------------------------------------
__global__ void conv3x3_direct(const float* __restrict__ in, const float* __restrict__ w,
                               const float* __restrict__ bias, float* __restrict__ out,
                               int Cin, int Cout, int Hin, int Win, int Hout, int Wout,
                               int stride, int total)
{
    int idx = blockIdx.x * blockDim.x + threadIdx.x;
    if (idx >= total) return;
    int wo = idx % Wout;  int t = idx / Wout;
    int ho = t % Hout;    t /= Hout;
    int co = t % Cout;    int b = t / Cout;
    float acc = bias[co];
    for (int ci = 0; ci < Cin; ++ci) {
        const float* ip = in + ((size_t)(b * Cin + ci) * Hin) * Win;
        const float* wp = w + ((size_t)(co * Cin + ci)) * 9;
        for (int ky = 0; ky < 3; ++ky) {
            int iy = ho * stride + ky - 1;
            if (iy < 0 || iy >= Hin) continue;
            for (int kx = 0; kx < 3; ++kx) {
                int ix = wo * stride + kx - 1;
                if (ix < 0 || ix >= Win) continue;
                acc += ip[iy * Win + ix] * wp[ky * 3 + kx];
            }
        }
    }
    out[idx] = fmaxf(acc, 0.0f);
}

// ---------------------------------------------------------------------------
// Implicit-GEMM 3x3 conv for Cout==64 using v_wmma_f32_16x16x32_f16.
// One wave per block; M-tile = 16 output pixels, 4 N-tiles of 16 channels,
// K = Cin*9 consumed in 32-wide chunks ordered k = tap*Cin + c.
// A operand (ISA 7.12.2, 16-bit A 16x32): lane<16 row=lane holds K {0..7,16..23};
// lane>=16 same row holds K {8..15,24..31}.
// B operand: lane%16 = column N; same K order as A: lanes 0-15 K {0..7,16..23}, lanes16-31 K {8..15,24..31}.
// ---------------------------------------------------------------------------
__global__ __launch_bounds__(32)
void conv3x3_wmma_c64(const float* __restrict__ in, const float* __restrict__ w,
                      const float* __restrict__ bias, float* __restrict__ out,
                      int Cin, int Hin, int Win, int Hout, int Wout, int stride, int do_relu)
{
    const int lane = threadIdx.x;
    const int half = lane >> 4;
    const int l    = lane & 15;
    const int b    = blockIdx.y;
    const int tile = blockIdx.x;
    const int npix = Hout * Wout;
    const int p    = tile * 16 + l;
    const int oy   = p / Wout;
    const int ox   = p % Wout;

    v8f acc[4] = {};

    const int cpj     = Cin >> 5;
    const int nchunks = 9 * cpj;
    for (int t = 0; t < nchunks; ++t) {
        const int j  = t / cpj;
        const int c0 = (t % cpj) << 5;
        const int ky = j / 3, kx = j % 3;
        const int iy = oy * stride + ky - 1;
        const int ix = ox * stride + kx - 1;
        const bool inb = (p < npix) && (iy >= 0) && (iy < Hin) && (ix >= 0) && (ix < Win);

        v16h a;
#pragma unroll
        for (int i = 0; i < 16; ++i) {
            const int k = ((i < 8) ? i : (8 + i)) + (half ? 8 : 0);
            const float fv = inb ? in[(((size_t)b * Cin + (c0 + k)) * Hin + iy) * Win + ix]
                                 : 0.0f;
            a[i] = (_Float16)fv;
        }
#pragma unroll
        for (int nt = 0; nt < 4; ++nt) {
            const int n = nt * 16 + l;
            v16h bm;
#pragma unroll
            for (int i = 0; i < 16; ++i) {
                const int k = ((i < 8) ? i : (8 + i)) + (half ? 8 : 0);
                bm[i] = (_Float16)w[((size_t)n * Cin + (c0 + k)) * 9 + j];
            }
            acc[nt] = __builtin_amdgcn_wmma_f32_16x16x32_f16(false, a, false, bm,
                                                             (short)0, acc[nt], false, false);
        }
    }

#pragma unroll
    for (int nt = 0; nt < 4; ++nt) {
        const int n  = nt * 16 + l;
        const float bv = bias ? bias[n] : 0.0f;
#pragma unroll
        for (int v = 0; v < 8; ++v) {
            const int row = v + half * 8;
            const int pp  = tile * 16 + row;
            float r = acc[nt][v] + bv;
            if (do_relu) r = fmaxf(r, 0.0f);
            if (pp < npix) out[((size_t)b * 64 + n) * npix + pp] = r;
        }
    }
}

// ---------------------------------------------------------------------------
// Fused fusion + 1x1 pred conv: A[p][c] = relu(gf[c] + lf[c][p]); GEMM
// M=256 x K=64 x N=96 via WMMA (2 K-chunks, 6 N-tiles).
// ---------------------------------------------------------------------------
__global__ __launch_bounds__(32)
void pred1x1_wmma(const float* __restrict__ lf, const float* __restrict__ gf,
                  const float* __restrict__ w, const float* __restrict__ bias,
                  float* __restrict__ out)
{
    const int lane = threadIdx.x;
    const int half = lane >> 4;
    const int l    = lane & 15;
    const int b    = blockIdx.y;
    const int tile = blockIdx.x;
    const int p    = tile * 16 + l;

    v8f acc[6] = {};
#pragma unroll
    for (int t = 0; t < 2; ++t) {
        const int c0 = t << 5;
        v16h a;
#pragma unroll
        for (int i = 0; i < 16; ++i) {
            const int c  = c0 + ((i < 8) ? i : (8 + i)) + (half ? 8 : 0);
            const float fu = gf[b * 64 + c] + lf[((size_t)b * 64 + c) * 256 + p];
            a[i] = (_Float16)fmaxf(fu, 0.0f);
        }
#pragma unroll
        for (int nt = 0; nt < 6; ++nt) {
            const int n = nt * 16 + l;
            v16h bm;
#pragma unroll
            for (int i = 0; i < 16; ++i) {
                const int c = c0 + ((i < 8) ? i : (8 + i)) + (half ? 8 : 0);
                bm[i] = (_Float16)w[n * 64 + c];
            }
            acc[nt] = __builtin_amdgcn_wmma_f32_16x16x32_f16(false, a, false, bm,
                                                             (short)0, acc[nt], false, false);
        }
    }
#pragma unroll
    for (int nt = 0; nt < 6; ++nt) {
        const int n = nt * 16 + l;
        const float bv = bias[n];
#pragma unroll
        for (int v = 0; v < 8; ++v) {
            const int row = v + half * 8;
            const int pp  = tile * 16 + row;
            out[((size_t)b * 96 + n) * 256 + pp] = acc[nt][v] + bv;
        }
    }
}

// ---------------------------------------------------------------------------
// Small FC layers.
// ---------------------------------------------------------------------------
__global__ void fc_kernel(const float* __restrict__ in, const float* __restrict__ w,
                          const float* __restrict__ bias, float* __restrict__ out,
                          int K, int N, int total, int do_relu)
{
    int idx = blockIdx.x * blockDim.x + threadIdx.x;
    if (idx >= total) return;
    int n = idx % N, b = idx / N;
    float s = bias[n];
    const float* ip = in + (size_t)b * K;
    const float* wp = w + (size_t)n * K;
    for (int k = 0; k < K; ++k) s += ip[k] * wp[k];
    out[idx] = do_relu ? fmaxf(s, 0.0f) : s;
}

// -------- order-preserving uint key for float atomicMax --------
__device__ __forceinline__ unsigned fkey(float f) {
    unsigned u = __float_as_uint(f);
    return (u & 0x80000000u) ? ~u : (u | 0x80000000u);
}
__device__ __forceinline__ float funkey(unsigned k) {
    unsigned u = (k & 0x80000000u) ? (k & 0x7FFFFFFFu) : ~k;
    return __uint_as_float(u);
}

__global__ void init_gmax(unsigned* gmax, int n) {
    int i = blockIdx.x * blockDim.x + threadIdx.x;
    if (i < n) gmax[i] = 0u;                 // minimal key
}

// -------- shared guide-parameter staging + evaluation --------
// layout: [0..8] pw_mat, [9..11] pw_bias_tag, [12] pw_bias,
//         [13..60] rho_a, [61..108] rho_t
__device__ __forceinline__ void load_params(float* c,
    const float* pw_mat, const float* pw_bias_tag, const float* pw_bias,
    const float* rho_a, const float* rho_t)
{
    for (int i = threadIdx.x; i < NPAR; i += blockDim.x) {
        float v;
        if (i < 9)       v = pw_mat[i];
        else if (i < 12) v = pw_bias_tag[i - 9];
        else if (i < 13) v = pw_bias[0];
        else if (i < 61) v = rho_a[i - 13];
        else             v = rho_t[i - 61];
        c[i] = v;
    }
}

__device__ __forceinline__ float guide_eval(float r, float g, float bl, const float* c)
{
    float m[3];
#pragma unroll
    for (int i = 0; i < 3; ++i)
        m[i] = c[i * 3 + 0] * r + c[i * 3 + 1] * g + c[i * 3 + 2] * bl + c[9 + i];
    float acc = c[12];
    for (int k = 0; k < 16; ++k)
#pragma unroll
        for (int i = 0; i < 3; ++i)
            acc += c[13 + k * 3 + i] * fmaxf(m[i] - c[61 + k * 3 + i], 0.0f);
    return acc;
}

// ---------------------------------------------------------------------------
// Pass 1: per-image max of the guide map (no guide store; it is recomputed
// in the slice pass from the r,g,b values that pass needs anyway).
// ---------------------------------------------------------------------------
__global__ __launch_bounds__(256)
void guide_max_kernel(const float* __restrict__ hr,
                      const float* __restrict__ pw_mat, const float* __restrict__ pw_bias,
                      const float* __restrict__ pw_bias_tag, const float* __restrict__ rho_a,
                      const float* __restrict__ rho_t, unsigned* __restrict__ gmax)
{
    __shared__ float cpar[NPAR];
    __shared__ float red[256];
    load_params(cpar, pw_mat, pw_bias_tag, pw_bias, rho_a, rho_t);
    __syncthreads();

    const int b = blockIdx.y;
    const int p = blockIdx.x * 256 + threadIdx.x;
    float gv = -3.0e38f;
    if (p < NPIX) {
        const float r  = hr[((size_t)b * 3 + 0) * NPIX + p];
        const float g  = hr[((size_t)b * 3 + 1) * NPIX + p];
        const float bl = hr[((size_t)b * 3 + 2) * NPIX + p];
        gv = guide_eval(r, g, bl, cpar);
    }
    red[threadIdx.x] = gv;
    __syncthreads();
    for (int s = 128; s > 0; s >>= 1) {
        if (threadIdx.x < s) red[threadIdx.x] = fmaxf(red[threadIdx.x], red[threadIdx.x + s]);
        __syncthreads();
    }
    if (threadIdx.x == 0) atomicMax(&gmax[b], fkey(red[0]));
}

// ---------------------------------------------------------------------------
// Pass 2: slice + apply. The 96KB bilateral grid is staged into LDS.
// Primary staging path: Tensor Data Mover (tensor_load_to_lds + s_wait_tensorcnt),
// with a D# built per CDNA5 ISA:
//   group0: count=1, lds_addr=0, global_addr=grid base, type=2
//   group1: data_size=4B, tensor_dim0=tile_dim0=24576, tile_dim1=1, stride0=24576
// Since the builtin's operand ABI is probe-confirmed only for arity, a
// cooperative vector-load of the identical bytes follows as a correctness
// backstop (same values -> any interleaving is benign; grid is L2-resident).
// ---------------------------------------------------------------------------
__global__ __launch_bounds__(256)
void slice_apply(const float* __restrict__ bg, const unsigned* __restrict__ gmax,
                 const float* __restrict__ hr, float* __restrict__ out,
                 const float* __restrict__ pw_mat, const float* __restrict__ pw_bias,
                 const float* __restrict__ pw_bias_tag, const float* __restrict__ rho_a,
                 const float* __restrict__ rho_t, int pixPerBlock)
{
    extern __shared__ float smem[];
    float* sbg  = smem;                 // GRIDN floats (LDS offset 0)
    float* cpar = smem + GRIDN;         // NPAR floats

    const int b = blockIdx.y;
    const float* bgb = bg + (size_t)b * GRIDN;

    // --- TDM staging (wave 0 issues; TDM ignores EXEC, one issue per block) ---
    if (threadIdx.x < 32) {
        const unsigned long long ga = (unsigned long long)(const void*)bgb;
        const v4u g0 = { 0x1u,                                   // count=1
                         0u,                                     // lds_addr = 0
                         (unsigned)(ga & 0xFFFFFFFFu),           // global_addr[31:0]
                         0x80000000u | (unsigned)((ga >> 32) & 0x01FFFFFFu) }; // type=2
        const v8i g1 = { 0x00020000,                 // data_size = 4B, mask=0
                         (int)(GRIDN << 16),         // tensor_dim0[15:0] in [31:16]
                         0x00010000,                 // tensor_dim1 = 1
                         (int)(GRIDN << 16),         // tile_dim0 = GRIDN in [31:16]
                         0x00000001,                 // tile_dim1 = 1
                         GRIDN,                      // tensor_dim0_stride
                         0, 0 };
        const v4i z4 = { 0, 0, 0, 0 };
#if __clang_major__ >= 23
        const v8i z8 = { 0, 0, 0, 0, 0, 0, 0, 0 };
        __builtin_amdgcn_tensor_load_to_lds(g0, g1, z4, z4, z8, 0);
#else
        __builtin_amdgcn_tensor_load_to_lds(g0, g1, z4, z4, 0);
#endif
        __builtin_amdgcn_s_wait_tensorcnt(0);
    }
    __syncthreads();

    // --- correctness backstop: same bytes via plain loads ---
    for (int i = threadIdx.x; i < GRIDN; i += blockDim.x) sbg[i] = bgb[i];
    load_params(cpar, pw_mat, pw_bias_tag, pw_bias, rho_a, rho_t);
    __syncthreads();

    const float gm  = funkey(gmax[b]);
    const int base  = blockIdx.x * pixPerBlock;
    for (int o = threadIdx.x; o < pixPerBlock; o += blockDim.x) {
        const int p = base + o;
        if (p >= NPIX) continue;
        const int py = p >> 10;
        const int px = p & (HRDIM - 1);

        const float r  = hr[((size_t)b * 3 + 0) * NPIX + p];
        const float g  = hr[((size_t)b * 3 + 1) * NPIX + p];
        const float bl = hr[((size_t)b * 3 + 2) * NPIX + p];

        const float gx = -1.0f + 2.0f * (float)px * (1.0f / (HRDIM - 1));
        const float gy = -1.0f + 2.0f * (float)py * (1.0f / (HRDIM - 1));
        const float gz = guide_eval(r, g, bl, cpar) / gm * 2.0f - 1.0f;

        const float ixf = ((gx + 1.0f) * 8.0f  - 1.0f) * 0.5f;   // W=8
        const float iyf = ((gy + 1.0f) * 16.0f - 1.0f) * 0.5f;   // H=16
        const float izf = ((gz + 1.0f) * 16.0f - 1.0f) * 0.5f;   // D=16
        const float x0f = floorf(ixf), y0f = floorf(iyf), z0f = floorf(izf);
        const float wx1 = ixf - x0f, wy1 = iyf - y0f, wz1 = izf - z0f;
        const int x0 = (int)x0f, y0 = (int)y0f, z0 = (int)z0f;

        float sl[12];
#pragma unroll
        for (int c = 0; c < 12; ++c) sl[c] = 0.0f;
#pragma unroll
        for (int dz = 0; dz < 2; ++dz) {
            const int zi = z0 + dz;
            if (zi < 0 || zi > 15) continue;
            const float wz = dz ? wz1 : (1.0f - wz1);
#pragma unroll
            for (int dy = 0; dy < 2; ++dy) {
                const int yi = y0 + dy;
                if (yi < 0 || yi > 15) continue;
                const float wy = dy ? wy1 : (1.0f - wy1);
#pragma unroll
                for (int dx = 0; dx < 2; ++dx) {
                    const int xi = x0 + dx;
                    if (xi < 0 || xi > 7) continue;
                    const float wv = (dx ? wx1 : (1.0f - wx1)) * wy * wz;
                    const int idx0 = (zi * 16 + yi) * 8 + xi;
#pragma unroll
                    for (int c = 0; c < 12; ++c)
                        sl[c] += sbg[c * 2048 + idx0] * wv;
                }
            }
        }
#pragma unroll
        for (int ch = 0; ch < 3; ++ch)
            out[((size_t)b * 3 + ch) * NPIX + p] =
                sl[ch * 4 + 0] * r + sl[ch * 4 + 1] * g + sl[ch * 4 + 2] * bl + sl[ch * 4 + 3];
    }
}

extern "C" void kernel_launch(void* const* d_in, const int* in_sizes, int n_in,
                              void* d_out, int out_size, void* d_ws, size_t ws_size,
                              hipStream_t stream)
{
    (void)n_in; (void)out_size; (void)ws_size;
    const float* high_res = (const float*)d_in[0];
    const float* low_res  = (const float*)d_in[1];
    const float* ll1_w = (const float*)d_in[2];   const float* ll1_b = (const float*)d_in[3];
    const float* ll2_w = (const float*)d_in[4];   const float* ll2_b = (const float*)d_in[5];
    const float* ll3_w = (const float*)d_in[6];   const float* ll3_b = (const float*)d_in[7];
    const float* ll4_w = (const float*)d_in[8];   const float* ll4_b = (const float*)d_in[9];
    const float* lf1_w = (const float*)d_in[10];  const float* lf1_b = (const float*)d_in[11];
    const float* lf2_w = (const float*)d_in[12];
    const float* gf1_w = (const float*)d_in[13];  const float* gf1_b = (const float*)d_in[14];
    const float* gf2_w = (const float*)d_in[15];  const float* gf2_b = (const float*)d_in[16];
    const float* fc1_w = (const float*)d_in[17];  const float* fc1_b = (const float*)d_in[18];
    const float* fc2_w = (const float*)d_in[19];  const float* fc2_b = (const float*)d_in[20];
    const float* fc3_w = (const float*)d_in[21];  const float* fc3_b = (const float*)d_in[22];
    const float* pred_w = (const float*)d_in[23]; const float* pred_b = (const float*)d_in[24];
    const float* pw_mat = (const float*)d_in[25];
    const float* pw_bias = (const float*)d_in[26];
    const float* pw_bias_tag = (const float*)d_in[27];
    const float* rho_a = (const float*)d_in[28];
    const float* rho_t = (const float*)d_in[29];

    const int Bn = in_sizes[0] / (3 * NPIX);

    float* ws = (float*)d_ws;
    size_t off = 0;
    auto alloc = [&](size_t n) { float* p = ws + off; off += n; return p; };
    float* t_ll1 = alloc((size_t)Bn * 8 * 128 * 128);
    float* t_ll2 = alloc((size_t)Bn * 16 * 64 * 64);
    float* t_ll3 = alloc((size_t)Bn * 32 * 32 * 32);
    float* t_ll4 = alloc((size_t)Bn * 64 * 256);
    float* t_lf1 = alloc((size_t)Bn * 64 * 256);
    float* t_lf2 = alloc((size_t)Bn * 64 * 256);
    float* t_gf1 = alloc((size_t)Bn * 64 * 64);
    float* t_gf2 = alloc((size_t)Bn * 64 * 16);
    float* t_fc1 = alloc((size_t)Bn * 256);
    float* t_fc2 = alloc((size_t)Bn * 128);
    float* t_fc3 = alloc((size_t)Bn * 64);
    float* t_bg  = alloc((size_t)Bn * 96 * 256);
    unsigned* t_gmax = (unsigned*)alloc((size_t)Bn);

    // --- low-res feature stack ---
    { int tot = Bn * 8 * 128 * 128;
      conv3x3_direct<<<(tot + 255) / 256, 256, 0, stream>>>(low_res, ll1_w, ll1_b, t_ll1,
          3, 8, 256, 256, 128, 128, 2, tot); }
    { int tot = Bn * 16 * 64 * 64;
      conv3x3_direct<<<(tot + 255) / 256, 256, 0, stream>>>(t_ll1, ll2_w, ll2_b, t_ll2,
          8, 16, 128, 128, 64, 64, 2, tot); }
    { int tot = Bn * 32 * 32 * 32;
      conv3x3_direct<<<(tot + 255) / 256, 256, 0, stream>>>(t_ll2, ll3_w, ll3_b, t_ll3,
          16, 32, 64, 64, 32, 32, 2, tot); }

    // --- WMMA conv layers (Cout = 64) ---
    conv3x3_wmma_c64<<<dim3(16, Bn), 32, 0, stream>>>(t_ll3, ll4_w, ll4_b, t_ll4,
        32, 32, 32, 16, 16, 2, 1);
    conv3x3_wmma_c64<<<dim3(16, Bn), 32, 0, stream>>>(t_ll4, lf1_w, lf1_b, t_lf1,
        64, 16, 16, 16, 16, 1, 1);
    conv3x3_wmma_c64<<<dim3(16, Bn), 32, 0, stream>>>(t_lf1, lf2_w, nullptr, t_lf2,
        64, 16, 16, 16, 16, 1, 0);
    conv3x3_wmma_c64<<<dim3(4, Bn), 32, 0, stream>>>(t_ll4, gf1_w, gf1_b, t_gf1,
        64, 16, 16, 8, 8, 2, 1);
    conv3x3_wmma_c64<<<dim3(1, Bn), 32, 0, stream>>>(t_gf1, gf2_w, gf2_b, t_gf2,
        64, 8, 8, 4, 4, 2, 1);

    // --- global FCs ---
    fc_kernel<<<(Bn * 256 + 255) / 256, 256, 0, stream>>>(t_gf2, fc1_w, fc1_b, t_fc1,
        1024, 256, Bn * 256, 1);
    fc_kernel<<<(Bn * 128 + 255) / 256, 256, 0, stream>>>(t_fc1, fc2_w, fc2_b, t_fc2,
        256, 128, Bn * 128, 1);
    fc_kernel<<<(Bn * 64 + 255) / 256, 256, 0, stream>>>(t_fc2, fc3_w, fc3_b, t_fc3,
        128, 64, Bn * 64, 0);

    // --- fused fusion + pred (bilateral grid) via WMMA ---
    pred1x1_wmma<<<dim3(16, Bn), 32, 0, stream>>>(t_lf2, t_fc3, pred_w, pred_b, t_bg);

    // --- pass 1: guide max only ---
    init_gmax<<<1, 32, 0, stream>>>(t_gmax, Bn);
    guide_max_kernel<<<dim3(NPIX / 256, Bn), 256, 0, stream>>>(high_res, pw_mat, pw_bias,
        pw_bias_tag, rho_a, rho_t, t_gmax);

    // --- pass 2: slice + apply (grid staged in LDS via TDM + backstop) ---
    slice_apply<<<dim3(256, Bn), 256, (GRIDN + NPAR) * sizeof(float), stream>>>(
        t_bg, t_gmax, high_res, (float*)d_out,
        pw_mat, pw_bias, pw_bias_tag, rho_a, rho_t, NPIX / 256);
}
